// FixMemoryAdaptiveUpdatewithPA_88596585382836
// MI455X (gfx1250) — matrix operand order repaired and checked by reading, not compiled, not run on hardware
//
#include <hip/hip_runtime.h>

typedef __attribute__((ext_vector_type(16))) _Float16 v16h;
typedef __attribute__((ext_vector_type(8)))  float    v8f;

#define B_  32
#define CF_ 512
#define CC_ 256
#define M_  256
#define N_  4096
#define K2_ 128

__device__ __forceinline__ v8f wmma16x16x32(v16h a, v16h b, v8f c) {
  // D = A(16x32 f16) x B(32x16 f16) + C(16x16 f32)
  return __builtin_amdgcn_wmma_f32_16x16x32_f16(false, a, false, b, (short)0, c, false, false);
}

// ---- CDNA5 async global->LDS copy helpers (ASYNCcnt-tracked) --------------
__device__ __forceinline__ unsigned ldsOffsetOf(const void* p) {
  // generic pointers to LDS carry the workgroup-relative offset in the low 32 bits
  return (unsigned)(size_t)p;
}
__device__ __forceinline__ void asyncCopy16(unsigned lds_off, const void* gptr) {
  unsigned long long ga = (unsigned long long)gptr;
  asm volatile("global_load_async_to_lds_b128 %0, %1, off"
               :: "v"(lds_off), "v"(ga) : "memory");
}
__device__ __forceinline__ void waitAsync0() {
#if __has_builtin(__builtin_amdgcn_s_wait_asynccnt)
  __builtin_amdgcn_s_wait_asynccnt(0);
#else
  asm volatile("s_wait_asynccnt 0x0" ::: "memory");
#endif
}

__device__ __forceinline__ float waveSum(float v) {
#pragma unroll
  for (int o = 16; o > 0; o >>= 1) v += __shfl_xor(v, o, 32);
  return v;
}
__device__ __forceinline__ float waveMax(float v) {
#pragma unroll
  for (int o = 16; o > 0; o >>= 1) v = fmaxf(v, __shfl_xor(v, o, 32));
  return v;
}
__device__ __forceinline__ float blockSum256(float v, float* sh) {
  int t = threadIdx.x;
  sh[t] = v; __syncthreads();
  for (int s = 128; s > 0; s >>= 1) { if (t < s) sh[t] += sh[t + s]; __syncthreads(); }
  float r = sh[0]; __syncthreads();
  return r;
}
// first-index-on-tie argmax over 256 values (matches jnp.argmax); all threads get result
__device__ __forceinline__ int argmax256(float v, float* red, int* ired) {
  int t = threadIdx.x;
  red[t] = v; ired[t] = t; __syncthreads();
  for (int s = 128; s > 0; s >>= 1) {
    if (t < s && (red[t + s] > red[t] || (red[t + s] == red[t] && ired[t + s] < ired[t]))) {
      red[t] = red[t + s]; ired[t] = ired[t + s];
    }
    __syncthreads();
  }
  int r = ired[0]; __syncthreads();
  return r;
}

// ---------------------------------------------------------------------------
// Kernel 1: x = proj_w @ feats + proj_b   (per batch: [256x512]@[512x4096])
// B tile is staged TRANSPOSED ([n][k]) so each lane's 16 fragment halves are
// contiguous -> 2x ds_load_b128 instead of 16x ds_load_u16.
// ---------------------------------------------------------------------------
__global__ __launch_bounds__(256) void proj_gemm_kernel(
    const float* __restrict__ feats, const float* __restrict__ proj_w,
    const float* __restrict__ proj_b, float* __restrict__ out) {
  const int n0 = blockIdx.x * 128;
  const int c0 = blockIdx.y * 64;
  const int b  = blockIdx.z;
  __shared__ _Float16 As[64][40];    // [c][k] tile of proj_w (f16), 16B-aligned rows
  __shared__ _Float16 Bt[128][40];   // [n][k] transposed feats tile (f16)
  __shared__ float    Cs[64][128];   // output staging for coalesced store
  const int t = threadIdx.x;
  const int lane = t & 31, w = t >> 5;
  const int msub = w & 3, nq = w >> 2;       // msub 0..3, nq 0..1
  v8f acc[4];
#pragma unroll
  for (int j = 0; j < 4; ++j)
#pragma unroll
    for (int r = 0; r < 8; ++r) acc[j][r] = 0.0f;

  for (int k0 = 0; k0 < CF_; k0 += 32) {
#pragma unroll
    for (int r = 0; r < 8; ++r) {            // A: 64x32 = 2048 elems
      int idx = t + r * 256;
      int cl = idx >> 5, kl = idx & 31;
      As[cl][kl] = (_Float16)proj_w[(size_t)(c0 + cl) * CF_ + k0 + kl];
    }
#pragma unroll
    for (int r = 0; r < 4; ++r) {            // B: 32x128 floats, transpose into [n][k]
      int idx = t + r * 256;                 // 0..1023 float4s
      int fl = idx >> 5;                     // k row 0..31
      int n4 = (idx & 31) * 4;               // n col group
      const float4 v = *(const float4*)(feats + ((size_t)b * CF_ + k0 + fl) * N_ + n0 + n4);
      Bt[n4 + 0][fl] = (_Float16)v.x;
      Bt[n4 + 1][fl] = (_Float16)v.y;
      Bt[n4 + 2][fl] = (_Float16)v.z;
      Bt[n4 + 3][fl] = (_Float16)v.w;
    }
    __syncthreads();
    v16h a;
#pragma unroll
    for (int i = 0; i < 16; ++i)
      a[i] = As[msub * 16 + (lane & 15)][(lane >> 4) * 16 + i];
#pragma unroll
    for (int j = 0; j < 4; ++j) {
      const _Float16* bp = &Bt[nq * 64 + j * 16 + (lane & 15)][(lane >> 4) * 16];
      v16h bf;
#pragma unroll
      for (int i = 0; i < 16; ++i) bf[i] = bp[i];   // contiguous 32B -> 2x ds_load_b128
      acc[j] = wmma16x16x32(a, bf, acc[j]);
    }
    __syncthreads();
  }
#pragma unroll
  for (int j = 0; j < 4; ++j)
#pragma unroll
    for (int r = 0; r < 8; ++r) {
      int mrow = msub * 16 + r + 8 * (lane >> 4);
      Cs[mrow][nq * 64 + j * 16 + (lane & 15)] = acc[j][r];
    }
  __syncthreads();
#pragma unroll
  for (int r = 0; r < 32; ++r) {
    int idx = t + r * 256;
    int row = idx >> 7, col = idx & 127;
    out[((size_t)b * (2 * CC_) + c0 + row) * N_ + n0 + col] = Cs[row][col] + proj_b[c0 + row];
  }
}

// ---------------------------------------------------------------------------
// Kernel 2: pooled[b][c] = mean_n( x[b][c][n] * preds[b][n] )
// ---------------------------------------------------------------------------
__global__ __launch_bounds__(256) void pooled_kernel(
    const float* __restrict__ out, const float* __restrict__ preds,
    float* __restrict__ pooled) {
  __shared__ float red[256];
  int bid = blockIdx.x;
  int b = bid >> 8, c = bid & 255;
  int t = threadIdx.x;
  const float* xrow = out + ((size_t)b * (2 * CC_) + c) * N_;
  const float* prow = preds + (size_t)b * N_;
  float s = 0.0f;
  for (int n = t; n < N_; n += 256) s += xrow[n] * prow[n];
  float tot = blockSum256(s, red);
  if (t == 0) pooled[b * CC_ + c] = tot * (1.0f / N_);
}

// ---------------------------------------------------------------------------
// Kernel 3: sequential memory EMA update (faithful to the scan). Single block.
// ---------------------------------------------------------------------------
__global__ __launch_bounds__(256) void memupd_kernel(
    float* mem, const float* __restrict__ pooled) {
  __shared__ float vsh[256], red[256], logits[256], norms[256], rowv[256];
  __shared__ int   ired[256];
  __shared__ float s_rate;
  const int t = threadIdx.x;
  for (int i = 0; i < B_; ++i) {
    vsh[t] = pooled[i * CC_ + t];
    __syncthreads();
    float vnorm = fmaxf(sqrtf(blockSum256(vsh[t] * vsh[t], red)), 1e-12f);
    float ss = 0.0f, dv = 0.0f;
    const float* mrow = mem + (size_t)t * CC_;
    for (int c = 0; c < CC_; ++c) { float m = mrow[c]; ss += m * m; dv += m * vsh[c]; }
    float nm = fmaxf(sqrtf(ss), 1e-12f);
    norms[t]  = nm;
    logits[t] = dv / (nm * vnorm);
    __syncthreads();
    int idx = argmax256(logits[t], red, ired);
    float val_i = logits[idx];
    const float* krow = mem + (size_t)idx * CC_;
    float d2 = 0.0f;
    for (int c = 0; c < CC_; ++c) d2 += mrow[c] * krow[c];
    rowv[t] = d2 / (norms[t] * norms[idx]);
    __syncthreads();
    int i1 = argmax256(rowv[t], red, ired);                          // top-1
    int hard = argmax256((t == i1) ? -3.0e38f : rowv[t], red, ired); // top-2
    if (t == 0) {
      float val_q = logits[hard];
      s_rate = val_q / (val_q + val_i);
    }
    __syncthreads();
    float rate = s_rate;
    mem[(size_t)idx * CC_ + t] = mem[(size_t)idx * CC_ + t] * rate + (1.0f - rate) * vsh[t];
    __syncthreads();
  }
}

// ---------------------------------------------------------------------------
// Kernel 4: per-row L2 norm of updated memory -> f16 normalized [M][C] and
// f16 transposed raw [C][M] for the two attention GEMMs.
// ---------------------------------------------------------------------------
__global__ __launch_bounds__(256) void memhalf_kernel(
    const float* __restrict__ mem, _Float16* __restrict__ memn_h,
    _Float16* __restrict__ memT_h) {
  __shared__ float red[256];
  int m = blockIdx.x, c = threadIdx.x;
  float v = mem[(size_t)m * CC_ + c];
  float norm = fmaxf(sqrtf(blockSum256(v * v, red)), 1e-12f);
  memn_h[(size_t)m * CC_ + c] = (_Float16)(v / norm);
  memT_h[(size_t)c * M_ + m] = (_Float16)v;
}

// ---------------------------------------------------------------------------
// Kernel 5: fused attention over a 32-pixel tile of one batch.
//   x tile is pulled into LDS with CDNA5 async copies (ASYNCcnt), then
//   transposed/converted to f16; S = l2n(x)@memn^T; P = softmax_M(S);
//   R = P@mem; writes R (pre-gate) + per-pixel gate logit.
// ---------------------------------------------------------------------------
__global__ __launch_bounds__(256) void attn_kernel(
    float* out,                               // d_out: read x, write read-tile
    const _Float16* __restrict__ memn_h,      // [M][C] normalized
    const _Float16* __restrict__ memT_h,      // [C][M] raw, transposed
    const float* __restrict__ gq_w, const float* __restrict__ gq_b,
    float* __restrict__ sgate) {
  const int n0 = blockIdx.x * 32;
  const int b  = blockIdx.y;
  __shared__ __attribute__((aligned(16)))
      union { float Xraw[CC_][32]; float Sb[32][M_]; } uA;          // 32 KB
  __shared__ __attribute__((aligned(16)))
      union { _Float16 Xh[32][CC_]; _Float16 P[32][M_]; } uB;       // 16 KB
  __shared__ float rnorm[32];
  const int t = threadIdx.x, lane = t & 31, w = t >> 5;

  // Phase 0a: async copy raw f32 x tile [c][j] straight from global (no reformat)
  {
    const float* grow = out + ((size_t)b * (2 * CC_) + t) * N_ + n0;  // 128B row
    unsigned lrow = ldsOffsetOf(&uA.Xraw[t][0]);
#pragma unroll
    for (int q = 0; q < 8; ++q)
      asyncCopy16(lrow + q * 16, grow + q * 4);
  }
  waitAsync0();
  __syncthreads();
  // Phase 0b: transpose + convert to f16 [j][c]
  {
#pragma unroll
    for (int j = 0; j < 32; ++j)
      uB.Xh[j][t] = (_Float16)uA.Xraw[t][j];
  }
  __syncthreads();
  // per-pixel 1/||x|| (folded into logits after the GEMM: row scaling commutes)
  for (int j = w * 4; j < w * 4 + 4; ++j) {
    float ss = 0.0f;
    for (int c = lane; c < CC_; c += 32) { float x = (float)uB.Xh[j][c]; ss += x * x; }
    ss = waveSum(ss);
    if (lane == 0) rnorm[j] = 1.0f / fmaxf(sqrtf(ss), 1e-12f);
  }
  __syncthreads();

  const int nsub = w & 1, mq = w >> 1;
  const int no = nsub * 16;
  v8f acc[4];
#pragma unroll
  for (int j = 0; j < 4; ++j)
#pragma unroll
    for (int r = 0; r < 8; ++r) acc[j][r] = 0.0f;

  // Phase A: S = X @ memn^T   (K = C = 256); Sb overlays the now-dead Xraw
  for (int k0 = 0; k0 < CC_; k0 += 32) {
    v16h a;
#pragma unroll
    for (int i = 0; i < 16; ++i)
      a[i] = uB.Xh[no + (lane & 15)][k0 + (lane >> 4) * 16 + i];
#pragma unroll
    for (int j = 0; j < 4; ++j) {
      const _Float16* bp = memn_h + (size_t)(mq * 64 + j * 16 + (lane & 15)) * CC_
                                  + k0 + (lane >> 4) * 16;
      v16h bf;
#pragma unroll
      for (int i = 0; i < 16; ++i) bf[i] = bp[i];   // contiguous 32B global
      acc[j] = wmma16x16x32(a, bf, acc[j]);
    }
  }
#pragma unroll
  for (int j = 0; j < 4; ++j)
#pragma unroll
    for (int r = 0; r < 8; ++r) {
      int pr = no + r + 8 * (lane >> 4);
      uA.Sb[pr][mq * 64 + j * 16 + (lane & 15)] = acc[j][r] * rnorm[pr];
    }
  __syncthreads();

  // Phase B: softmax over M per pixel -> P (f16, overlays Xh which is now dead)
  for (int j = w * 4; j < w * 4 + 4; ++j) {
    float mx = -3.0e38f;
    for (int m = lane; m < M_; m += 32) mx = fmaxf(mx, uA.Sb[j][m]);
    mx = waveMax(mx);
    float sm = 0.0f;
    for (int m = lane; m < M_; m += 32) sm += __expf(uA.Sb[j][m] - mx);
    sm = waveSum(sm);
    float inv = 1.0f / sm;
    for (int m = lane; m < M_; m += 32) uB.P[j][m] = (_Float16)(__expf(uA.Sb[j][m] - mx) * inv);
  }
  __syncthreads();

  // Phase C: R = P @ mem   (K = M = 256)
#pragma unroll
  for (int j = 0; j < 4; ++j)
#pragma unroll
    for (int r = 0; r < 8; ++r) acc[j][r] = 0.0f;
  for (int k0 = 0; k0 < M_; k0 += 32) {
    v16h a;
#pragma unroll
    for (int i = 0; i < 16; ++i)
      a[i] = uB.P[no + (lane & 15)][k0 + (lane >> 4) * 16 + i];
#pragma unroll
    for (int j = 0; j < 4; ++j) {
      const _Float16* bp = memT_h + (size_t)(mq * 64 + j * 16 + (lane & 15)) * M_
                                  + k0 + (lane >> 4) * 16;
      v16h bf;
#pragma unroll
      for (int i = 0; i < 16; ++i) bf[i] = bp[i];
      acc[j] = wmma16x16x32(a, bf, acc[j]);
    }
  }
  __syncthreads();
#pragma unroll
  for (int j = 0; j < 4; ++j)
#pragma unroll
    for (int r = 0; r < 8; ++r) {
      int pr = no + r + 8 * (lane >> 4);
      uA.Sb[pr][mq * 64 + j * 16 + (lane & 15)] = acc[j][r];   // Sb now holds R
    }
  __syncthreads();

  // Phase D: per-pixel gate logit + store R (pre-gate) to concat second half
  for (int j = w * 4; j < w * 4 + 4; ++j) {
    float s = 0.0f;
    for (int c = lane; c < CC_; c += 32) s += gq_w[c] * uA.Sb[j][c];
    s = waveSum(s);
    if (lane == 0) sgate[(size_t)b * N_ + n0 + j] = s + gq_b[0];
  }
  for (int j = 0; j < 32; ++j)
    out[((size_t)b * (2 * CC_) + CC_ + t) * N_ + n0 + j] = uA.Sb[j][t];
}

// ---------------------------------------------------------------------------
// Kernel 6: per-batch gating path: q = softmax_N(sgate); rbar = sum_n q*read;
// z = gv_w@rbar + gv_b; gate = sigmoid(LN(z@wz_w^T + wz_b))
// sgate row pulled in with async copies.
// ---------------------------------------------------------------------------
__global__ __launch_bounds__(256) void gate_kernel(
    const float* __restrict__ outb, const float* __restrict__ sgate,
    const float* __restrict__ gv_w, const float* __restrict__ gv_b,
    const float* __restrict__ wz_w, const float* __restrict__ wz_b,
    const float* __restrict__ ln_g, const float* __restrict__ ln_b,
    float* __restrict__ gate) {
  __shared__ __attribute__((aligned(16))) float q[N_];   // 16 KB
  __shared__ float red[256];
  __shared__ float rb[CC_];
  __shared__ float zs[K2_];
  const int b = blockIdx.x, t = threadIdx.x;
  {
    const float* grow = sgate + (size_t)b * N_ + t * 16;  // 64B per thread
    unsigned lrow = ldsOffsetOf(&q[t * 16]);
#pragma unroll
    for (int i = 0; i < 4; ++i)
      asyncCopy16(lrow + i * 16, grow + i * 4);
  }
  waitAsync0();
  __syncthreads();
  float mx = -3.0e38f;
  for (int n = t; n < N_; n += 256) mx = fmaxf(mx, q[n]);
  red[t] = mx; __syncthreads();
  for (int s = 128; s > 0; s >>= 1) { if (t < s) red[t] = fmaxf(red[t], red[t + s]); __syncthreads(); }
  mx = red[0]; __syncthreads();
  float sm = 0.0f;
  for (int n = t; n < N_; n += 256) { float e = __expf(q[n] - mx); q[n] = e; sm += e; }
  float inv = 1.0f / blockSum256(sm, red);
  __syncthreads();
  {
    const float* rrow = outb + ((size_t)b * (2 * CC_) + CC_ + t) * N_;
    float s = 0.0f;
    for (int n = 0; n < N_; ++n) s += q[n] * rrow[n];
    rb[t] = s * inv;
  }
  __syncthreads();
  if (t < K2_) {
    float s = gv_b[t];
    for (int c = 0; c < CC_; ++c) s += gv_w[t * CC_ + c] * rb[c];
    zs[t] = s;
  }
  __syncthreads();
  float tv = wz_b[t];
  for (int k = 0; k < K2_; ++k) tv += wz_w[t * K2_ + k] * zs[k];
  float mu  = blockSum256(tv, red) * (1.0f / CC_);
  float dvv = tv - mu;
  float var = blockSum256(dvv * dvv, red) * (1.0f / CC_);
  float g = dvv * rsqrtf(var + 1e-5f) * ln_g[t] + ln_b[t];
  gate[b * CC_ + t] = 1.0f / (1.0f + __expf(-g));
}

// ---------------------------------------------------------------------------
// Kernel 7: in-place gating of the second concat half
// ---------------------------------------------------------------------------
__global__ __launch_bounds__(256) void apply_gate_kernel(
    float* __restrict__ outb, const float* __restrict__ gate) {
  int bid = blockIdx.x;
  int b = bid >> 8, c = bid & 255;
  float g = gate[b * CC_ + c];
  float* row = outb + ((size_t)b * (2 * CC_) + CC_ + c) * N_;
  for (int n = threadIdx.x; n < N_; n += 256) row[n] *= g;
}

extern "C" void kernel_launch(void* const* d_in, const int* in_sizes, int n_in,
                              void* d_out, int out_size, void* d_ws, size_t ws_size,
                              hipStream_t stream) {
  (void)in_sizes; (void)n_in; (void)out_size; (void)ws_size;
  const float* feats  = (const float*)d_in[0];
  const float* preds  = (const float*)d_in[1];
  const float* proj_w = (const float*)d_in[2];
  const float* proj_b = (const float*)d_in[3];
  const float* memory = (const float*)d_in[4];
  const float* gq_w   = (const float*)d_in[5];
  const float* gq_b   = (const float*)d_in[6];
  const float* gv_w   = (const float*)d_in[7];
  const float* gv_b   = (const float*)d_in[8];
  const float* wz_w   = (const float*)d_in[9];
  const float* wz_b   = (const float*)d_in[10];
  const float* ln_g   = (const float*)d_in[11];
  const float* ln_b   = (const float*)d_in[12];
  float* out = (float*)d_out;

  char* ws = (char*)d_ws;
  float*    mem_f32 = (float*)ws;    ws += (size_t)M_ * CC_ * 4;
  float*    pooled  = (float*)ws;    ws += (size_t)B_ * CC_ * 4;
  float*    sgate   = (float*)ws;    ws += (size_t)B_ * N_  * 4;
  float*    gate    = (float*)ws;    ws += (size_t)B_ * CC_ * 4;
  _Float16* memn_h  = (_Float16*)ws; ws += (size_t)M_ * CC_ * 2;
  _Float16* memT_h  = (_Float16*)ws; ws += (size_t)M_ * CC_ * 2;

  proj_gemm_kernel<<<dim3(N_ / 128, CC_ / 64, B_), 256, 0, stream>>>(feats, proj_w, proj_b, out);
  pooled_kernel<<<dim3(B_ * CC_), 256, 0, stream>>>(out, preds, pooled);
  hipMemcpyAsync(mem_f32, memory, (size_t)M_ * CC_ * 4, hipMemcpyDeviceToDevice, stream);
  memupd_kernel<<<dim3(1), 256, 0, stream>>>(mem_f32, pooled);
  memhalf_kernel<<<dim3(M_), 256, 0, stream>>>(mem_f32, memn_h, memT_h);
  attn_kernel<<<dim3(N_ / 32, B_), 256, 0, stream>>>(out, memn_h, memT_h, gq_w, gq_b, sgate);
  gate_kernel<<<dim3(B_), 256, 0, stream>>>(out, sgate, gv_w, gv_b, wz_w, wz_b, ln_g, ln_b, gate);
  apply_gate_kernel<<<dim3(B_ * CC_), 256, 0, stream>>>(out, gate);
}
